// StructuredStateSpaceDualityBranch_55379308315187
// MI455X (gfx1250) — matrix operand order repaired and checked by reading, not compiled, not run on hardware
//
#include <hip/hip_runtime.h>
#include <hip/hip_bf16.h>
#include <math.h>
#include <stdint.h>

// ---------------- problem constants ----------------
#define B_   4
#define L_   2048
#define DM_  1024
#define DI_  2048
#define DS_  64
#define DC_  4
#define H_   16
#define DTR_ 64
#define P_   128           // DI/H
#define TOTAL_ 6208        // 2*DI + DTR + 2*H*DS
#define ROWS_ (B_*L_)      // 8192
// column offsets inside the in_proj output "p"
#define OFF_Z   0
#define OFF_U   DI_            // 2048
#define OFF_DTH (2*DI_)        // 4096
#define OFF_B   (2*DI_+DTR_)   // 4160
#define OFF_C   (2*DI_+DTR_+H_*DS_) // 5184

typedef __attribute__((ext_vector_type(16))) __bf16 v16bf;
typedef __attribute__((ext_vector_type(8)))  __bf16 v8bf;
typedef __attribute__((ext_vector_type(8)))  float  v8f;

// ---------------- f32 -> bf16 cast ----------------
__global__ void cast_f32_bf16(const float* __restrict__ in, __bf16* __restrict__ out, long n) {
  long i = (long)blockIdx.x * blockDim.x + threadIdx.x;
  long stride = (long)gridDim.x * blockDim.x;
  for (; i < n; i += stride) out[i] = (__bf16)in[i];
}

// ---------------- bf16 WMMA GEMM: C[M,N] = A[M,K] * W[N,K]^T ----------------
// Block: 256 threads (8 waves). Block tile: 256(M) x 64(N). Wave tile: 64x32
// (4 M-frags x 2 N-frags = 8 WMMA accumulators per wave).
// A tile (256x32 bf16, 16KB) staged into LDS with GLOBAL_LOAD_ASYNC_TO_LDS_B128
// (ASYNCcnt-tracked async copy, no VGPR round-trip). W fragments are loaded
// directly from global: all bf16 weights (<21MB) are L2-resident on the 192MB L2.
__global__ __launch_bounds__(256)
void gemm_bf16_wmma(const __bf16* __restrict__ A, const __bf16* __restrict__ W,
                    float* __restrict__ C, int M, int N, int K)
{
  __shared__ alignas(32) __bf16 sA[256 * 32];   // 16 KB

  const int tid  = threadIdx.x;
  const int lane = tid & 31;
  const int wid  = tid >> 5;      // 0..7
  const int wm   = wid & 3;       // 64-row strip within the 256-row block tile
  const int wn   = wid >> 2;      // 32-col strip within the 64-col block tile
  const int half = lane >> 4;     // 0/1
  const int l16  = lane & 15;

  const int m0 = blockIdx.y * 256;
  const int n0 = blockIdx.x * 64;

  v8f acc[4][2] = {};

  // Async staging: thread tid owns one 64-byte row slice of the A tile.
  // Generic LDS pointers are {aperture_hi, lds_offset}: low 32 bits = LDS byte addr.
  const unsigned ldsRow = (unsigned)(uintptr_t)(&sA[tid * 32]);
  const __bf16* gaBase = A + (size_t)(m0 + tid) * K;

  for (int ks = 0; ks < K; ks += 32) {
    __syncthreads();                 // previous iteration's LDS reads done
    {
      const __bf16* ga = gaBase + ks;
      // offset applies to both LDS and global addresses (ISA 15.18.3)
      asm volatile("global_load_async_to_lds_b128 %0, %1, off"           :: "v"(ldsRow), "v"(ga) : "memory");
      asm volatile("global_load_async_to_lds_b128 %0, %1, off offset:16" :: "v"(ldsRow), "v"(ga) : "memory");
      asm volatile("global_load_async_to_lds_b128 %0, %1, off offset:32" :: "v"(ldsRow), "v"(ga) : "memory");
      asm volatile("global_load_async_to_lds_b128 %0, %1, off offset:48" :: "v"(ldsRow), "v"(ga) : "memory");
      asm volatile("s_wait_asynccnt 0x0" ::: "memory");
    }
    __syncthreads();

    // B fragments straight from global (row n contiguous in K)
    v16bf bfr[2];
#pragma unroll
    for (int ni = 0; ni < 2; ++ni) {
      const int n = n0 + wn * 32 + ni * 16 + l16;
      const __bf16* gb = W + (size_t)n * K + ks + half * 16;
      bfr[ni] = *(const v16bf*)gb;
      if (ks + 32 < K) __builtin_prefetch(gb + 32, 0, 1);   // global_prefetch_b8
    }

#pragma unroll
    for (int mi = 0; mi < 4; ++mi) {
      const int m = wm * 64 + mi * 16 + l16;
      // A frag contiguous layout: K in {half*16 .. +15}
      v8bf lo = *(const v8bf*)&sA[m * 32 + half * 16];
      v8bf hi = *(const v8bf*)&sA[m * 32 + half * 16 + 8];
      v16bf af = __builtin_shufflevector(lo, hi, 0,1,2,3,4,5,6,7,8,9,10,11,12,13,14,15);
#pragma unroll
      for (int ni = 0; ni < 2; ++ni)
        acc[mi][ni] = __builtin_amdgcn_wmma_f32_16x16x32_bf16(
            false, af, false, bfr[ni], (short)0, acc[mi][ni], false, false);
    }
  }

  // D layout: VGPR r -> M = r + 8*half, N = l16
#pragma unroll
  for (int mi = 0; mi < 4; ++mi)
#pragma unroll
    for (int ni = 0; ni < 2; ++ni) {
      const int col = n0 + wn * 32 + ni * 16 + l16;
      const int rowbase = m0 + wm * 64 + mi * 16 + half * 8;
#pragma unroll
      for (int r = 0; r < 8; ++r)
        C[(size_t)(rowbase + r) * N + col] = acc[mi][ni][r];
    }
}

// ---------------- causal depthwise conv (DC=4) ----------------
__global__ void conv_kernel(const float* __restrict__ p, const float* __restrict__ cw,
                            const float* __restrict__ cb, float* __restrict__ uo)
{
  long idx = (long)blockIdx.x * blockDim.x + threadIdx.x;   // over ROWS_*DI_
  if (idx >= (long)ROWS_ * DI_) return;
  const int  c  = (int)(idx & (DI_ - 1));
  const long bt = idx >> 11;                // row index b*L + t
  const int  t  = (int)(bt & (L_ - 1));
  float acc = cb[c];
#pragma unroll
  for (int j = 0; j < DC_; ++j) {
    const int tt = t - (DC_ - 1) + j;
    if (tt >= 0)
      acc += p[(size_t)(bt - (DC_ - 1 - j)) * TOTAL_ + OFF_U + c] * cw[c * DC_ + j];
  }
  uo[idx] = acc;
}

// ---------------- dt = clip(softplus(dth @ dtw^T + bias)) ----------------
__global__ void dt_kernel(const float* __restrict__ p, const float* __restrict__ dtw,
                          const float* __restrict__ dtb, float* __restrict__ dt)
{
  const int idx = blockIdx.x * blockDim.x + threadIdx.x;    // ROWS_*H_
  if (idx >= ROWS_ * H_) return;
  const int h = idx & (H_ - 1);
  const int row = idx >> 4;
  const float* dh = p + (size_t)row * TOTAL_ + OFF_DTH;
  float acc = dtb[h];
#pragma unroll 8
  for (int k = 0; k < DTR_; ++k) acc += dh[k] * dtw[h * DTR_ + k];
  const float sp = (acc > 20.f) ? acc : log1pf(expf(acc));
  dt[idx] = fminf(fmaxf(sp, 1e-4f), 1.0f);
}

// ---------------- sequential SSD scan: one block per (b,h), thread = row p ----------------
__global__ __launch_bounds__(128)
void scan_kernel(const float* __restrict__ p, const float* __restrict__ dt,
                 const float* __restrict__ A_log, const float* __restrict__ Dskip,
                 float* __restrict__ uy /* in: conv(u), out: y (in-place) */)
{
  const int b  = blockIdx.x >> 4;
  const int h  = blockIdx.x & (H_ - 1);
  const int pi = threadIdx.x;           // 0..127

  __shared__ float sDec[DS_], sDtB[DS_], sC[DS_];

  float An = 0.f;
  if (pi < DS_) An = -__expf(A_log[h * DS_ + pi]);
  const float Dh = Dskip[h * P_ + pi];

  float hst[DS_];
#pragma unroll
  for (int n = 0; n < DS_; ++n) hst[n] = 0.f;

  for (int t = 0; t < L_; ++t) {
    const size_t row = (size_t)b * L_ + t;
    const float dtv = dt[row * H_ + h];
    if (pi < DS_) {
      sDec[pi] = __expf(dtv * An);
      sDtB[pi] = dtv * p[row * TOTAL_ + OFF_B + h * DS_ + pi];
      sC[pi]   =        p[row * TOTAL_ + OFF_C + h * DS_ + pi];
    }
    __syncthreads();
    const size_t ui = row * DI_ + h * P_ + pi;
    const float uv = uy[ui];
    float acc = Dh * uv;
#pragma unroll
    for (int n = 0; n < DS_; ++n) {
      const float hh = sDec[n] * hst[n] + sDtB[n] * uv;
      hst[n] = hh;
      acc += hh * sC[n];
    }
    uy[ui] = acc;
    __syncthreads();
  }
}

// ---------------- fuse: g = y*silu(z) + res ; RMSNorm ; -> bf16 ----------------
__global__ __launch_bounds__(256)
void fuse_norm_kernel(const float* __restrict__ y, const float* __restrict__ p,
                      const float* __restrict__ res, const float* __restrict__ nw,
                      __bf16* __restrict__ g16)
{
  const int row = blockIdx.x;                       // ROWS_
  const float* zrow = p + (size_t)row * TOTAL_ + OFF_Z;
  const float* yrow = y + (size_t)row * DI_;
  const float* rrow = res + (size_t)row * DI_;

  float gv[8];
  float ss = 0.f;
#pragma unroll
  for (int j = 0; j < 8; ++j) {
    const int i = threadIdx.x + j * 256;
    const float z = zrow[i];
    const float g = yrow[i] * (z / (1.f + __expf(-z))) + rrow[i];
    gv[j] = g;
    ss += g * g;
  }
  // wave32 reduce + cross-wave via LDS
#pragma unroll
  for (int m = 16; m >= 1; m >>= 1) ss += __shfl_xor(ss, m);
  __shared__ float red[8];
  const int lane = threadIdx.x & 31, wv = threadIdx.x >> 5;
  if (lane == 0) red[wv] = ss;
  __syncthreads();
  if (threadIdx.x == 0) {
    float t = 0.f;
#pragma unroll
    for (int w = 0; w < 8; ++w) t += red[w];
    red[0] = t;
  }
  __syncthreads();
  const float rms = rsqrtf(red[0] / (float)DI_ + 1e-6f);
#pragma unroll
  for (int j = 0; j < 8; ++j) {
    const int i = threadIdx.x + j * 256;
    g16[(size_t)row * DI_ + i] = (__bf16)(gv[j] * rms * nw[i]);
  }
}

// ---------------- host-side launch ----------------
extern "C" void kernel_launch(void* const* d_in, const int* in_sizes, int n_in,
                              void* d_out, int out_size, void* d_ws, size_t ws_size,
                              hipStream_t stream) {
  const float* x        = (const float*)d_in[0];   // [B,L,DM]
  const float* in_w     = (const float*)d_in[1];   // [6208,1024]
  const float* dt_w     = (const float*)d_in[2];   // [16,64]
  const float* conv_w   = (const float*)d_in[3];   // [2048,4]
  const float* conv_b   = (const float*)d_in[4];   // [2048]
  const float* A_log    = (const float*)d_in[5];   // [16,64]
  const float* Dskip    = (const float*)d_in[6];   // [16,128]
  const float* dt_bias  = (const float*)d_in[7];   // [16]
  const float* norm_w   = (const float*)d_in[8];   // [2048]
  const float* out_w    = (const float*)d_in[9];   // [1024,2048]
  const float* res_w    = (const float*)d_in[10];  // [2048,1024]
  float* out = (float*)d_out;                      // [B,L,DM]

  // workspace carve-out (256B aligned)
  char* w = (char*)d_ws;
  auto alloc = [&](size_t bytes) -> void* {
    void* r = (void*)w;
    w += (bytes + 255) & ~(size_t)255;
    return r;
  };
  __bf16* x16    = (__bf16*)alloc((size_t)ROWS_ * DM_ * 2);       // 16 MB
  __bf16* win16  = (__bf16*)alloc((size_t)TOTAL_ * DM_ * 2);      // 12.7 MB
  __bf16* wres16 = (__bf16*)alloc((size_t)DI_ * DM_ * 2);         // 4 MB
  __bf16* wout16 = (__bf16*)alloc((size_t)DM_ * DI_ * 2);         // 4 MB
  float*  p      = (float*)alloc((size_t)ROWS_ * TOTAL_ * 4);     // 203 MB
  float*  uy     = (float*)alloc((size_t)ROWS_ * DI_ * 4);        // 67 MB (conv out, then y in-place)
  float*  dt     = (float*)alloc((size_t)ROWS_ * H_ * 4);         // 0.5 MB
  float*  res    = (float*)alloc((size_t)ROWS_ * DI_ * 4);        // 67 MB
  __bf16* g16    = (__bf16*)alloc((size_t)ROWS_ * DI_ * 2);       // 33 MB
  (void)ws_size; (void)n_in; (void)in_sizes; (void)out_size;

  // 1) bf16 casts (activations + weights; weights become L2-resident)
  {
    long n;
    n = (long)ROWS_ * DM_;
    cast_f32_bf16<<<dim3((unsigned)((n + 255) / 256)), dim3(256), 0, stream>>>(x, x16, n);
    n = (long)TOTAL_ * DM_;
    cast_f32_bf16<<<dim3((unsigned)((n + 255) / 256)), dim3(256), 0, stream>>>(in_w, win16, n);
    n = (long)DI_ * DM_;
    cast_f32_bf16<<<dim3((unsigned)((n + 255) / 256)), dim3(256), 0, stream>>>(res_w, wres16, n);
    n = (long)DM_ * DI_;
    cast_f32_bf16<<<dim3((unsigned)((n + 255) / 256)), dim3(256), 0, stream>>>(out_w, wout16, n);
  }

  // 2) p = x @ in_proj_w^T   (M=8192, N=6208, K=1024)
  gemm_bf16_wmma<<<dim3(TOTAL_ / 64, ROWS_ / 256), dim3(256), 0, stream>>>(
      x16, win16, p, ROWS_, TOTAL_, DM_);

  // 3) depthwise causal conv on the u slice of p -> uy
  {
    long n = (long)ROWS_ * DI_;
    conv_kernel<<<dim3((unsigned)((n + 255) / 256)), dim3(256), 0, stream>>>(p, conv_w, conv_b, uy);
  }

  // 4) dt projection + softplus + clip
  dt_kernel<<<dim3((ROWS_ * H_ + 255) / 256), dim3(256), 0, stream>>>(p, dt_w, dt_bias, dt);

  // 5) residual_inner = x @ res_proj_w^T  (M=8192, N=2048, K=1024)
  gemm_bf16_wmma<<<dim3(DI_ / 64, ROWS_ / 256), dim3(256), 0, stream>>>(
      x16, wres16, res, ROWS_, DI_, DM_);

  // 6) sequential SSD scan: y written in-place over uy
  scan_kernel<<<dim3(B_ * H_), dim3(128), 0, stream>>>(p, dt, A_log, Dskip, uy);

  // 7) gate + residual + RMSNorm -> bf16
  fuse_norm_kernel<<<dim3(ROWS_), dim3(256), 0, stream>>>(uy, p, res, norm_w, g16);

  // 8) out = g @ out_proj_w^T  (M=8192, N=1024, K=2048)
  gemm_bf16_wmma<<<dim3(DM_ / 64, ROWS_ / 256), dim3(256), 0, stream>>>(
      g16, wout16, out, ROWS_, DM_, DI_);
}
